// Dilated_DConv_71382356459626
// MI455X (gfx1250) — compile-verified, matrix operand order checked by reading, not run on hardware
//
#include <hip/hip_runtime.h>
#include <hip/hip_bf16.h>

// Deformable conv for MI455X (gfx1250, wave32, WMMA).
//
// Stage 1: pack w_def and w_off (zero-padded M 18->32) into bf16 WMMA
//          A-fragment layouts in d_ws.
// Stage 2: offset = conv3x3(x, w_off) as a WMMA GEMM:
//          off(32 x Npix) = Woff(32x576) x im2col(x), K order kk*64+c.
// Stage 3: fused bilinear-sample + WMMA GEMM:
//          out(64 x Npix) = Wdef(64x576) x deform_im2col(x).

typedef __attribute__((ext_vector_type(16))) __bf16 v16bf;
typedef __attribute__((ext_vector_type(8)))  float  v8f;

#define BB 4
#define CC 64
#define HH 128
#define WW 128
#define HWHW (HH*WW)

#define NDEF (18 * 4 * 32 * 16)   // packed w_def entries (4 M-tiles)
#define NOFF (18 * 2 * 32 * 16)   // packed w_off entries (2 M-tiles)

// ---------------------------------------------------------------------------
// Kernel 1: pack both weight tensors to bf16 A fragments.
// K global = kk*64 + c.  Fragment layout per ISA 7.12.2 (16-bit A 16x32):
//   lanes 0-15  : K-locals {0..7, 16..23}
//   lanes 16-31 : K-locals {8..15, 24..31}
// Flat layout: apack[ks][mt][lane][16] bf16.
// ---------------------------------------------------------------------------
__global__ void __launch_bounds__(256)
pack_weights_kernel(const float* __restrict__ w_def,
                    const float* __restrict__ w_off,
                    __bf16* __restrict__ apackdef,
                    __bf16* __restrict__ apackoff)
{
    int tid = blockIdx.x * 256 + threadIdx.x;
    if (tid < NDEF) {
        int idx  = tid & 15;
        int lane = (tid >> 4) & 31;
        int mt   = (tid >> 9) & 3;
        int ks   = tid >> 11;
        int row    = mt * 16 + (lane & 15);          // output channel o (0..63)
        int klocal = ((idx < 8) ? 0 : 16) + ((lane >> 4) << 3) + (idx & 7);
        int Kg     = ks * 32 + klocal;
        int kk     = Kg >> 6;
        int c      = Kg & 63;
        apackdef[tid] = (__bf16)w_def[(size_t)(row * CC + c) * 9 + kk];
    } else if (tid < NDEF + NOFF) {
        int t2   = tid - NDEF;
        int idx  = t2 & 15;
        int lane = (t2 >> 4) & 31;
        int mt   = (t2 >> 9) & 1;
        int ks   = t2 >> 10;
        int row    = mt * 16 + (lane & 15);          // offset channel o (0..31)
        int klocal = ((idx < 8) ? 0 : 16) + ((lane >> 4) << 3) + (idx & 7);
        int Kg     = ks * 32 + klocal;
        int kk     = Kg >> 6;
        int c      = Kg & 63;
        apackoff[t2] = (row < 18)
            ? (__bf16)w_off[(size_t)(row * CC + c) * 9 + kk]
            : (__bf16)0.f;
    }
}

// ---------------------------------------------------------------------------
// Kernel 2: offset conv as WMMA GEMM.  Workgroup = 4 waves = 64 consecutive
// x pixels of one row (b, y).  Wave: 16 pixels (N) x 32 channels (M, 2 tiles).
// Each K-step has fixed kk -> the im2col column is a single shifted pixel,
// so the B fragment is 16 plain channel loads per lane.
// ---------------------------------------------------------------------------
__global__ void __launch_bounds__(128)
offset_wmma_kernel(const float* __restrict__ xin,
                   const __bf16* __restrict__ apackoff,
                   float* __restrict__ offs)
{
    int wg   = blockIdx.x;           // 0..1023
    int b    = wg >> 8;
    int rem  = wg & 255;
    int y    = rem >> 1;
    int xb0  = (rem & 1) << 6;
    int lane = threadIdx.x & 31;
    int wave = threadIdx.x >> 5;
    int p    = lane & 15;
    int hi   = lane >> 4;
    int xi   = xb0 + wave * 16 + p;

    const float* xbat = xin + (size_t)b * CC * HWHW;
    v8f acc0 = {}, acc1 = {};

    for (int ks = 0; ks < 18; ++ks) {
        int kk = ks >> 1;
        int c0 = (ks & 1) << 5;
        int ty = y  + kk / 3 - 1;
        int tx = xi + kk % 3 - 1;
        bool valid = (ty >= 0) && (ty < HH) && (tx >= 0) && (tx < WW);
        int   idx = valid ? (ty * WW + tx) : 0;
        float msk = valid ? 1.f : 0.f;

        v16bf bfrag;
#pragma unroll
        for (int t = 0; t < 16; ++t) {
            int c = c0 + ((t < 8) ? 0 : 16) + hi * 8 + (t & 7);
            bfrag[t] = (__bf16)(xbat[c * HWHW + idx] * msk);
        }

        const __bf16* ap = apackoff + ((size_t)(ks * 2 * 32) + lane) * 16;
        v16bf a0 = *(const v16bf*)(ap);
        v16bf a1 = *(const v16bf*)(ap + 512);

        acc0 = __builtin_amdgcn_wmma_f32_16x16x32_bf16(false, a0, false, bfrag,
                                                       (short)0, acc0, false, false);
        acc1 = __builtin_amdgcn_wmma_f32_16x16x32_bf16(false, a1, false, bfrag,
                                                       (short)0, acc1, false, false);
    }

    // C/D layout: VGPR r -> M = r (lanes 0-15) / r+8 (lanes 16-31).
    float* ob = offs + (size_t)b * 18 * HWHW + y * WW + xi;
#pragma unroll
    for (int r = 0; r < 8; ++r) {
        int m = r + hi * 8;
        ob[(size_t)m * HWHW] = acc0[r];
    }
    if (hi == 0) {                    // rows 16,17 of the padded 32; 18..31 dropped
        ob[(size_t)16 * HWHW] = acc1[0];
        ob[(size_t)17 * HWHW] = acc1[1];
    }
}

// ---------------------------------------------------------------------------
// Kernel 3: fused bilinear sampling + WMMA GEMM.
// Wave: 16 pixels (N), 64 output channels (M) -> 4x v8f accumulators.
// Each K-step has fixed kk -> bilinear coords computed once per lane and
// reused across the 16 channels it samples.
// ---------------------------------------------------------------------------
__global__ void __launch_bounds__(128)
deform_wmma_kernel(const float* __restrict__ xin,
                   const float* __restrict__ offs,
                   const __bf16* __restrict__ apack,
                   float* __restrict__ out)
{
    int wg   = blockIdx.x;           // 0..1023
    int b    = wg >> 8;
    int rem  = wg & 255;
    int y    = rem >> 1;
    int xb0  = (rem & 1) << 6;
    int lane = threadIdx.x & 31;
    int wave = threadIdx.x >> 5;
    int p    = lane & 15;
    int hi   = lane >> 4;
    int xi   = xb0 + wave * 16 + p;

    const float* xbat = xin  + (size_t)b * CC * HWHW;
    const float* offb = offs + (size_t)b * 18 * HWHW + y * WW + xi;

    v8f acc0 = {}, acc1 = {}, acc2 = {}, acc3 = {};

    for (int ks = 0; ks < 18; ++ks) {
        int kk = ks >> 1;
        int c0 = (ks & 1) << 5;

        float dy = offb[(kk * 2 + 0) * HWHW];
        float dx = offb[(kk * 2 + 1) * HWHW];
        float py  = (float)(y  + kk / 3 - 1) + dy;
        float pxx = (float)(xi + kk % 3 - 1) + dx;
        float fy = floorf(py), fx = floorf(pxx);
        float wy = py - fy,    wx = pxx - fx;
        int y0 = (int)fy, x0 = (int)fx;
        int y1 = y0 + 1,  x1 = x0 + 1;

        float m00 = (y0 >= 0 && y0 < HH && x0 >= 0 && x0 < WW) ? 1.f : 0.f;
        float m01 = (y0 >= 0 && y0 < HH && x1 >= 0 && x1 < WW) ? 1.f : 0.f;
        float m10 = (y1 >= 0 && y1 < HH && x0 >= 0 && x0 < WW) ? 1.f : 0.f;
        float m11 = (y1 >= 0 && y1 < HH && x1 >= 0 && x1 < WW) ? 1.f : 0.f;
        int cy0 = y0 < 0 ? 0 : (y0 > HH - 1 ? HH - 1 : y0);
        int cy1 = y1 < 0 ? 0 : (y1 > HH - 1 ? HH - 1 : y1);
        int cx0 = x0 < 0 ? 0 : (x0 > WW - 1 ? WW - 1 : x0);
        int cx1 = x1 < 0 ? 0 : (x1 > WW - 1 ? WW - 1 : x1);

        float w00 = (1.f - wy) * (1.f - wx) * m00;
        float w01 = (1.f - wy) * wx         * m01;
        float w10 = wy         * (1.f - wx) * m10;
        float w11 = wy         * wx         * m11;
        int i00 = cy0 * WW + cx0, i01 = cy0 * WW + cx1;
        int i10 = cy1 * WW + cx0, i11 = cy1 * WW + cx1;

        // B fragment (32x16 bf16, lane = N column, mirrored A layout).
        v16bf bfrag;
#pragma unroll
        for (int t = 0; t < 16; ++t) {
            int c = c0 + ((t < 8) ? 0 : 16) + hi * 8 + (t & 7);
            const float* xc = xbat + c * HWHW;
            float v = w00 * xc[i00] + w01 * xc[i01]
                    + w10 * xc[i10] + w11 * xc[i11];
            bfrag[t] = (__bf16)v;
        }

        // A fragments: contiguous 32B per lane, identical across waves (L1 hit).
        const __bf16* ap = apack + ((size_t)(ks * 4 * 32) + lane) * 16;
        v16bf a0 = *(const v16bf*)(ap);
        v16bf a1 = *(const v16bf*)(ap + 512);
        v16bf a2 = *(const v16bf*)(ap + 1024);
        v16bf a3 = *(const v16bf*)(ap + 1536);

        acc0 = __builtin_amdgcn_wmma_f32_16x16x32_bf16(false, a0, false, bfrag,
                                                       (short)0, acc0, false, false);
        acc1 = __builtin_amdgcn_wmma_f32_16x16x32_bf16(false, a1, false, bfrag,
                                                       (short)0, acc1, false, false);
        acc2 = __builtin_amdgcn_wmma_f32_16x16x32_bf16(false, a2, false, bfrag,
                                                       (short)0, acc2, false, false);
        acc3 = __builtin_amdgcn_wmma_f32_16x16x32_bf16(false, a3, false, bfrag,
                                                       (short)0, acc3, false, false);
    }

    // C/D layout: VGPR r -> M = r (lanes 0-15) / r+8 (lanes 16-31), N = lane&15.
    float* ob = out + (size_t)b * CC * HWHW + y * WW + xi;
#pragma unroll
    for (int r = 0; r < 8; ++r) {
        int m = r + hi * 8;
        ob[(size_t)( 0 + m) * HWHW] = acc0[r];
        ob[(size_t)(16 + m) * HWHW] = acc1[r];
        ob[(size_t)(32 + m) * HWHW] = acc2[r];
        ob[(size_t)(48 + m) * HWHW] = acc3[r];
    }
}

// ---------------------------------------------------------------------------
extern "C" void kernel_launch(void* const* d_in, const int* in_sizes, int n_in,
                              void* d_out, int out_size, void* d_ws, size_t ws_size,
                              hipStream_t stream)
{
    const float* x     = (const float*)d_in[0];
    const float* w_off = (const float*)d_in[1];
    const float* w_def = (const float*)d_in[2];
    float*       out   = (float*)d_out;

    // Workspace layout:
    //   [0, 4.718 MB)         offsets (4*18*128*128 f32)
    //   [+0, 73728 B)         packed w_def fragments (bf16)
    //   [+73728, +110592 B)   packed w_off fragments (bf16, M padded to 32)
    float*  offs     = (float*)d_ws;
    char*   wsbase   = (char*)d_ws + (size_t)BB * 18 * HWHW * sizeof(float);
    __bf16* apackdef = (__bf16*)wsbase;
    __bf16* apackoff = (__bf16*)(wsbase + (size_t)NDEF * sizeof(__bf16));

    pack_weights_kernel<<<(NDEF + NOFF + 255) / 256, 256, 0, stream>>>(
        w_def, w_off, apackdef, apackoff);
    offset_wmma_kernel<<<(BB * HWHW) / 64, 128, 0, stream>>>(x, apackoff, offs);
    deform_wmma_kernel<<<(BB * HWHW) / 64, 128, 0, stream>>>(x, offs, apackdef, out);
}
